// DifferentialAttentionBlock_14121852469515
// MI455X (gfx1250) — compile-verified, matrix operand order checked by reading, not run on hardware
//
#include <hip/hip_runtime.h>
#include <math.h>

// ---------------------------------------------------------------------------
// Types for CDNA5 WMMA (wave32): bf16 16x16x32, f32 accumulate
// ---------------------------------------------------------------------------
typedef __attribute__((ext_vector_type(8)))  float        v8f;
typedef __attribute__((ext_vector_type(16))) __bf16       v16bf;
typedef __attribute__((ext_vector_type(8)))  __bf16       v8bf;
typedef __attribute__((ext_vector_type(4)))  unsigned int u32x4;
typedef __attribute__((ext_vector_type(8)))  int          i32x8;
typedef __attribute__((ext_vector_type(4)))  int          i32x4;

static __device__ __forceinline__ v16bf cat16(v8bf lo, v8bf hi) {
  return __builtin_shufflevector(lo, hi, 0,1,2,3,4,5,6,7,8,9,10,11,12,13,14,15);
}

// Load one WMMA 16-bit A/B operand: per-lane two contiguous 8-element runs
// (K = half*8 .. +7 and K = 16+half*8 .. +7), base already offset by half*8.
static __device__ __forceinline__ v16bf ld_op(const __bf16* base) {
  v8bf lo = *(const v8bf*)(base);
  v8bf hi = *(const v8bf*)(base + 16);
  return cat16(lo, hi);
}

static __device__ __forceinline__ v8f wmma_bf16(v16bf a, v16bf b, v8f c) {
  return __builtin_amdgcn_wmma_f32_16x16x32_bf16(false, a, false, b, (short)0, c,
                                                 false, false);
}

// ---------------------------------------------------------------------------
// TDM: issue a 2D tensor-tile DMA (bf16 elements) global -> LDS.
// D# built per CDNA5 ISA ch.8: group0 {count, lds_addr, global_addr, type=2},
// group1 {data_size=2B, tensor_dim0/1 = tile_dim0/1, dim0_stride}.
// Tile is tile_d0 (contiguous, elements) x tile_d1 rows, row stride = stride0.
// 6-arg builtin form (clang-23 / therock headers).
// ---------------------------------------------------------------------------
static __device__ __forceinline__ void tdm_load_2d(unsigned lds_addr,
                                                   unsigned long long gaddr,
                                                   unsigned tile_d0,
                                                   unsigned tile_d1,
                                                   unsigned long long stride0)
{
  u32x4 g0;
  g0[0] = 1u;                                                 // count=1
  g0[1] = lds_addr;                                           // LDS byte addr
  g0[2] = (unsigned)gaddr;                                    // gaddr[31:0]
  g0[3] = ((unsigned)(gaddr >> 32) & 0x01FFFFFFu) | 0x80000000u; // [56:32]+type=2
  i32x8 g1;
  g1[0] = (int)(1u << 16);                                    // data_size=2B
  g1[1] = (int)((tile_d0 & 0xFFFFu) << 16);                   // tensor_dim0 lo
  g1[2] = (int)(((tile_d0 >> 16) & 0xFFFFu) |                 // tensor_dim0 hi
                ((tile_d1 & 0xFFFFu) << 16));                 // tensor_dim1 lo
  g1[3] = (int)(((tile_d1 >> 16) & 0xFFFFu) |                 // tensor_dim1 hi
                ((tile_d0 & 0xFFFFu) << 16));                 // tile_dim0
  g1[4] = (int)(tile_d1 & 0xFFFFu);                           // tile_dim1
  g1[5] = (int)(unsigned)stride0;                             // dim0_stride lo
  g1[6] = (int)((unsigned)(stride0 >> 32) & 0xFFFFu);         // dim0_stride hi
  g1[7] = 0;
  i32x4 z4 = {0, 0, 0, 0};
  i32x8 z8 = {0, 0, 0, 0, 0, 0, 0, 0};
  __builtin_amdgcn_tensor_load_to_lds(g0, g1, z4, z4, z8, 0);
}

// ---------------------------------------------------------------------------
// Elementwise fp32 -> bf16 (activations). n multiple of 1024.
// ---------------------------------------------------------------------------
__global__ __launch_bounds__(256)
void cvt_bf16(const float* __restrict__ in, __bf16* __restrict__ out)
{
  int base = (blockIdx.x * 256 + threadIdx.x) * 4;
  float4 v = *(const float4*)(in + base);
  out[base + 0] = (__bf16)v.x;
  out[base + 1] = (__bf16)v.y;
  out[base + 2] = (__bf16)v.z;
  out[base + 3] = (__bf16)v.w;
}

// ---------------------------------------------------------------------------
// W [K,N] fp32 -> Wt [N,K] bf16, LDS-tiled 32x32 transpose (32x8 threads).
// ---------------------------------------------------------------------------
__global__ __launch_bounds__(256)
void transpose_bf16(const float* __restrict__ W, __bf16* __restrict__ Wt,
                    int K, int N)
{
  __shared__ float tile[32][33];
  const int kx = blockIdx.x * 32, nx = blockIdx.y * 32;
  const int tx = threadIdx.x & 31, ty = threadIdx.x >> 5;
  #pragma unroll
  for (int i = 0; i < 32; i += 8)
    tile[ty + i][tx] = W[(size_t)(kx + ty + i) * N + nx + tx];
  __syncthreads();
  #pragma unroll
  for (int i = 0; i < 32; i += 8)
    Wt[(size_t)(nx + ty + i) * K + kx + tx] = (__bf16)tile[tx][ty + i];
}

// ---------------------------------------------------------------------------
// TDM-fed tiled GEMM:  out = A[MxK](bf16) @ Wt[NxK](bf16)^T + bias[N]
// BM=64 BN=64 BK=32, 256 threads = 8 waves (4 along M, 2 along N).
// Wave 0 drives the Tensor Data Mover (double-buffered LDS tiles); all waves
// gather WMMA operands with two ds_load_b128 each and issue bf16 WMMAs.
// MODE 0: fp32 row-major [M,N]      (final Wo GEMM -> d_out)
// MODE 1: bf16 [B,H,S,32]           (q1/q2/k1/k2 projections, N=512)
// MODE 2: bf16 [B,H,64,S] = V^T     (Wv projection, N=1024)
// ---------------------------------------------------------------------------
template<int MODE>
__global__ __launch_bounds__(256)
void gemm_wmma_tdm(const __bf16* __restrict__ A, const __bf16* __restrict__ Wt,
                   const float* __restrict__ bias, float* __restrict__ outF,
                   __bf16* __restrict__ outB, int M, int N, int K)
{
  __shared__ __bf16 Abuf[2][64 * 32];   // [row][k]
  __shared__ __bf16 Bbuf[2][64 * 32];   // [n][k]

  const int tid   = threadIdx.x;
  const int mBase = blockIdx.x * 64;
  const int nBase = blockIdx.y * 64;
  const int wv    = tid >> 5;
  const int lane  = tid & 31;
  const int mw    = wv & 3;
  const int nw    = wv >> 2;
  const int half  = lane >> 4;
  const int q     = lane & 15;

  const __bf16* aTile = A  + (size_t)mBase * K;
  const __bf16* wTile = Wt + (size_t)nBase * K;

  v8f acc0 = {}; v8f acc1 = {};
  const int nCh = K >> 5;

  // prologue: DMA chunk 0 into buffer 0
  if (tid < 32) {
    tdm_load_2d((unsigned)(uintptr_t)&Abuf[0][0],
                (unsigned long long)(uintptr_t)(aTile), 32u, 64u, (unsigned)K);
    tdm_load_2d((unsigned)(uintptr_t)&Bbuf[0][0],
                (unsigned long long)(uintptr_t)(wTile), 32u, 64u, (unsigned)K);
    __builtin_amdgcn_s_wait_tensorcnt(0);
  }
  __syncthreads();

  for (int c = 0; c < nCh; ++c) {
    const int cur = c & 1;
    // overlap: DMA chunk c+1 into the other buffer while computing on cur
    if (c + 1 < nCh && tid < 32) {
      const int kk = (c + 1) * 32;
      tdm_load_2d((unsigned)(uintptr_t)&Abuf[cur ^ 1][0],
                  (unsigned long long)(uintptr_t)(aTile + kk), 32u, 64u,
                  (unsigned)K);
      tdm_load_2d((unsigned)(uintptr_t)&Bbuf[cur ^ 1][0],
                  (unsigned long long)(uintptr_t)(wTile + kk), 32u, 64u,
                  (unsigned)K);
    }

    v16bf a  = ld_op(&Abuf[cur][(mw * 16 + q) * 32 + half * 8]);
    v16bf b0 = ld_op(&Bbuf[cur][(nw * 32 +      q) * 32 + half * 8]);
    v16bf b1 = ld_op(&Bbuf[cur][(nw * 32 + 16 + q) * 32 + half * 8]);
    acc0 = wmma_bf16(a, b0, acc0);
    acc1 = wmma_bf16(a, b1, acc1);

    if (c + 1 < nCh) {
      if (tid < 32) __builtin_amdgcn_s_wait_tensorcnt(0);
      __syncthreads();
    }
  }

  // epilogue: D layout lane = n + 16*(m>>3), vgpr r -> m = half*8 + r
  #pragma unroll
  for (int j = 0; j < 2; ++j) {
    v8f acc = j ? acc1 : acc0;
    int n = nBase + nw * 32 + j * 16 + q;
    float bv = bias[n];
    #pragma unroll
    for (int r = 0; r < 8; ++r) {
      int m = mBase + mw * 16 + half * 8 + r;
      float val = acc[r] + bv;
      if (MODE == 0) {
        outF[(size_t)m * N + n] = val;
      } else if (MODE == 1) {
        int bb = m >> 10, s = m & 1023;
        int hh = n >> 5,  d = n & 31;
        outB[(((size_t)(bb * 16 + hh)) * 1024 + s) * 32 + d] = (__bf16)val;
      } else {
        int bb = m >> 10, s = m & 1023;
        int hh = n >> 6,  d = n & 63;
        outB[(((size_t)(bb * 16 + hh)) * 64 + d) * 1024 + s] = (__bf16)val;
      }
    }
  }
}

// ---------------------------------------------------------------------------
// Fused differential attention, flash-style online softmax.
// One INDEPENDENT wave per (b, h, 16-query tile). 4 waves / block, no block
// barriers (wave-local LDS hazards closed with s_wait_dscnt).
// Per 32-key chunk: 4 score WMMAs + 8 accumulate WMMAs (out^T = V^T @ P^T).
// ---------------------------------------------------------------------------
__global__ __launch_bounds__(128)
void diff_attn(const __bf16* __restrict__ q1h, const __bf16* __restrict__ q2h,
               const __bf16* __restrict__ k1h, const __bf16* __restrict__ k2h,
               const __bf16* __restrict__ vth, __bf16* __restrict__ ctx,
               const float* __restrict__ lq1, const float* __restrict__ lk1,
               const float* __restrict__ lq2, const float* __restrict__ lk2)
{
  __shared__ __bf16 Pbuf[4][2][16 * 32];   // per-wave prob tiles [query][key]
  __shared__ float  cbuf[4][32];           // per-wave broadcast scratch

  const int lane  = threadIdx.x & 31;
  const int wib   = threadIdx.x >> 5;
  const int wid   = blockIdx.x * 4 + wib;
  const int b     = wid >> 10;
  const int h     = (wid >> 6) & 15;
  const int qt    = wid & 63;
  const int qBase = qt * 16;
  const int half  = lane >> 4;
  const int q     = lane & 15;

  const __bf16* q1p = q1h + ((size_t)(b * 16 + h)) * 1024 * 32;
  const __bf16* q2p = q2h + ((size_t)(b * 16 + h)) * 1024 * 32;
  const __bf16* k1p = k1h + ((size_t)(b * 16 + h)) * 1024 * 32;
  const __bf16* k2p = k2h + ((size_t)(b * 16 + h)) * 1024 * 32;
  const __bf16* vtp = vth + ((size_t)(b * 16 + h)) * 64 * 1024;

  v16bf q1a = ld_op(q1p + (size_t)(qBase + q) * 32 + half * 8);
  v16bf q2a = ld_op(q2p + (size_t)(qBase + q) * 32 + half * 8);

  float m1[8], l1[8], m2[8], l2[8];
  #pragma unroll
  for (int r = 0; r < 8; ++r) { m1[r] = m2[r] = -1e30f; l1[r] = l2[r] = 0.f; }
  v8f o1t[4], o2t[4];
  #pragma unroll
  for (int d = 0; d < 4; ++d) { o1t[d] = (v8f){}; o2t[d] = (v8f){}; }

  const float sc  = 0.125f;                      // 1/sqrt(64)
  const int   nCh = (qBase + 16 + 31) >> 5;      // causal chunk count

  for (int c = 0; c < nCh; ++c) {
    const int kc = c * 32;
    v8f z = {};
    v16bf kb10 = ld_op(k1p + (size_t)(kc +      q) * 32 + half * 8);
    v16bf kb11 = ld_op(k1p + (size_t)(kc + 16 + q) * 32 + half * 8);
    v16bf kb20 = ld_op(k2p + (size_t)(kc +      q) * 32 + half * 8);
    v16bf kb21 = ld_op(k2p + (size_t)(kc + 16 + q) * 32 + half * 8);
    v8f s1a = wmma_bf16(q1a, kb10, z);
    v8f s1b = wmma_bf16(q1a, kb11, z);
    v8f s2a = wmma_bf16(q2a, kb20, z);
    v8f s2b = wmma_bf16(q2a, kb21, z);

    float c1[8], c2[8];
    #pragma unroll
    for (int r = 0; r < 8; ++r) {
      const int  qi  = qBase + half * 8 + r;
      const bool ok0 = (kc +      q) <= qi;
      const bool ok1 = (kc + 16 + q) <= qi;
      float v1a = ok0 ? s1a[r] * sc : -1e9f;
      float v1b = ok1 ? s1b[r] * sc : -1e9f;
      float v2a = ok0 ? s2a[r] * sc : -1e9f;
      float v2b = ok1 ? s2b[r] * sc : -1e9f;

      // ---- stream 1: online softmax row update ----
      float mx = fmaxf(v1a, v1b);
      #pragma unroll
      for (int o = 1; o < 16; o <<= 1) mx = fmaxf(mx, __shfl_xor(mx, o, 32));
      float mn = fmaxf(m1[r], mx);
      float p0 = __expf(v1a - mn), p1 = __expf(v1b - mn);
      float sm = p0 + p1;
      #pragma unroll
      for (int o = 1; o < 16; o <<= 1) sm += __shfl_xor(sm, o, 32);
      c1[r] = __expf(m1[r] - mn);
      l1[r] = l1[r] * c1[r] + sm;
      m1[r] = mn;
      Pbuf[wib][0][(half * 8 + r) * 32 +      q] = (__bf16)p0;
      Pbuf[wib][0][(half * 8 + r) * 32 + 16 + q] = (__bf16)p1;

      // ---- stream 2 ----
      float mx2 = fmaxf(v2a, v2b);
      #pragma unroll
      for (int o = 1; o < 16; o <<= 1) mx2 = fmaxf(mx2, __shfl_xor(mx2, o, 32));
      float mn2 = fmaxf(m2[r], mx2);
      float p20 = __expf(v2a - mn2), p21 = __expf(v2b - mn2);
      float sm2 = p20 + p21;
      #pragma unroll
      for (int o = 1; o < 16; o <<= 1) sm2 += __shfl_xor(sm2, o, 32);
      c2[r] = __expf(m2[r] - mn2);
      l2[r] = l2[r] * c2[r] + sm2;
      m2[r] = mn2;
      Pbuf[wib][1][(half * 8 + r) * 32 +      q] = (__bf16)p20;
      Pbuf[wib][1][(half * 8 + r) * 32 + 16 + q] = (__bf16)p21;
    }

    if (q == 0) {
      #pragma unroll
      for (int r = 0; r < 8; ++r) {
        cbuf[wib][     half * 8 + r] = c1[r];
        cbuf[wib][16 + half * 8 + r] = c2[r];
      }
    }
    asm volatile("s_wait_dscnt 0x0" ::: "memory");
    const float c1q = cbuf[wib][q];
    const float c2q = cbuf[wib][16 + q];
    #pragma unroll
    for (int d = 0; d < 4; ++d) { o1t[d] *= c1q; o2t[d] *= c2q; }

    v16bf bp1 = ld_op(&Pbuf[wib][0][q * 32 + half * 8]);
    v16bf bp2 = ld_op(&Pbuf[wib][1][q * 32 + half * 8]);
    #pragma unroll
    for (int d = 0; d < 4; ++d) {
      v16bf av = ld_op(vtp + (size_t)(d * 16 + q) * 1024 + kc + half * 8);
      o1t[d] = wmma_bf16(av, bp1, o1t[d]);
      o2t[d] = wmma_bf16(av, bp2, o2t[d]);
    }
  }

  if (q == 0) {
    #pragma unroll
    for (int r = 0; r < 8; ++r) {
      cbuf[wib][     half * 8 + r] = l1[r];
      cbuf[wib][16 + half * 8 + r] = l2[r];
    }
  }
  asm volatile("s_wait_dscnt 0x0" ::: "memory");
  const float inv1 = 1.f / cbuf[wib][q];
  const float inv2 = 1.f / cbuf[wib][16 + q];
  const float lam  = __expf(lq1[0] * lk1[0]) - __expf(lq2[0] * lk2[0]) + 0.2f;

  __bf16* cp = ctx + ((size_t)(b * 1024) + qBase + q) * 1024 + h * 64;
  #pragma unroll
  for (int d = 0; d < 4; ++d)
    #pragma unroll
    for (int r = 0; r < 8; ++r)
      cp[d * 16 + half * 8 + r] =
          (__bf16)(o1t[d][r] * inv1 - lam * (o2t[d][r] * inv2));
}

// ---------------------------------------------------------------------------
// Launch: bf16 conversions + weight transposes -> 5 TDM GEMMs -> fused
// attention -> TDM output GEMM.
// ---------------------------------------------------------------------------
extern "C" void kernel_launch(void* const* d_in, const int* in_sizes, int n_in,
                              void* d_out, int out_size, void* d_ws, size_t ws_size,
                              hipStream_t stream)
{
  const float* q   = (const float*)d_in[0];
  const float* k   = (const float*)d_in[1];
  const float* v   = (const float*)d_in[2];
  // d_in[3] = mask (causal, computed analytically)
  const float* Wq1 = (const float*)d_in[4];  const float* bq1 = (const float*)d_in[5];
  const float* Wq2 = (const float*)d_in[6];  const float* bq2 = (const float*)d_in[7];
  const float* Wk1 = (const float*)d_in[8];  const float* bk1 = (const float*)d_in[9];
  const float* Wk2 = (const float*)d_in[10]; const float* bk2 = (const float*)d_in[11];
  const float* Wv  = (const float*)d_in[12]; const float* bv  = (const float*)d_in[13];
  const float* Wo  = (const float*)d_in[14]; const float* bo  = (const float*)d_in[15];
  const float* lq1 = (const float*)d_in[16]; const float* lk1 = (const float*)d_in[17];
  const float* lq2 = (const float*)d_in[18]; const float* lk2 = (const float*)d_in[19];

  char* ws = (char*)d_ws;
  const size_t MB = 1ull << 20;
  __bf16* qb   = (__bf16*)(ws + 0 * MB);    // [2048,1024] bf16, 4 MB
  __bf16* kb   = (__bf16*)(ws + 4 * MB);
  __bf16* vb   = (__bf16*)(ws + 8 * MB);
  __bf16* Wq1t = (__bf16*)(ws + 12 * MB);   // [512,1024]  bf16, 1 MB each
  __bf16* Wq2t = (__bf16*)(ws + 13 * MB);
  __bf16* Wk1t = (__bf16*)(ws + 14 * MB);
  __bf16* Wk2t = (__bf16*)(ws + 15 * MB);
  __bf16* Wvt  = (__bf16*)(ws + 16 * MB);   // [1024,1024] bf16, 2 MB
  __bf16* Wot  = (__bf16*)(ws + 18 * MB);
  __bf16* q1h  = (__bf16*)(ws + 20 * MB);   // [B,H,S,32]  bf16, 2 MB each
  __bf16* q2h  = (__bf16*)(ws + 22 * MB);
  __bf16* k1h  = (__bf16*)(ws + 24 * MB);
  __bf16* k2h  = (__bf16*)(ws + 26 * MB);
  __bf16* vth  = (__bf16*)(ws + 28 * MB);   // [B,H,64,S]  bf16, 4 MB
  __bf16* ctxb = (__bf16*)(ws + 32 * MB);   // [2048,1024] bf16, 4 MB

  dim3 blk(256);

  // activations fp32 -> bf16 (2M elements each, 4 per thread)
  cvt_bf16<<<2048, blk, 0, stream>>>(q, qb);
  cvt_bf16<<<2048, blk, 0, stream>>>(k, kb);
  cvt_bf16<<<2048, blk, 0, stream>>>(v, vb);

  // weights fp32 [K,N] -> bf16 [N,K]
  transpose_bf16<<<dim3(32, 16), blk, 0, stream>>>(Wq1, Wq1t, 1024, 512);
  transpose_bf16<<<dim3(32, 16), blk, 0, stream>>>(Wq2, Wq2t, 1024, 512);
  transpose_bf16<<<dim3(32, 16), blk, 0, stream>>>(Wk1, Wk1t, 1024, 512);
  transpose_bf16<<<dim3(32, 16), blk, 0, stream>>>(Wk2, Wk2t, 1024, 512);
  transpose_bf16<<<dim3(32, 32), blk, 0, stream>>>(Wv,  Wvt,  1024, 1024);
  transpose_bf16<<<dim3(32, 32), blk, 0, stream>>>(Wo,  Wot,  1024, 1024);

  dim3 gHalf(2048 / 64, 512 / 64);
  dim3 gFull(2048 / 64, 1024 / 64);
  gemm_wmma_tdm<1><<<gHalf, blk, 0, stream>>>(qb, Wq1t, bq1, nullptr, q1h, 2048, 512, 1024);
  gemm_wmma_tdm<1><<<gHalf, blk, 0, stream>>>(qb, Wq2t, bq2, nullptr, q2h, 2048, 512, 1024);
  gemm_wmma_tdm<1><<<gHalf, blk, 0, stream>>>(kb, Wk1t, bk1, nullptr, k1h, 2048, 512, 1024);
  gemm_wmma_tdm<1><<<gHalf, blk, 0, stream>>>(kb, Wk2t, bk2, nullptr, k2h, 2048, 512, 1024);
  gemm_wmma_tdm<2><<<gFull, blk, 0, stream>>>(vb, Wvt,  bv,  nullptr, vth, 2048, 1024, 1024);

  diff_attn<<<512, dim3(128), 0, stream>>>(q1h, q2h, k1h, k2h, vth, ctxb,
                                           lq1, lk1, lq2, lk2);

  gemm_wmma_tdm<0><<<gFull, blk, 0, stream>>>(ctxb, Wot, bo, (float*)d_out,
                                              nullptr, 2048, 1024, 1024);
}
